// Attention_14731737825621
// MI455X (gfx1250) — matrix operand order repaired and checked by reading, not compiled, not run on hardware
//
#include <hip/hip_runtime.h>

#define DEVINL __device__ __forceinline__

typedef __attribute__((ext_vector_type(16))) __bf16 bf16x16;
typedef __attribute__((ext_vector_type(8)))  float  f32x8;

constexpr int B_   = 2;
constexpr int S_   = 2048;
constexpr int E_   = 2048;
constexpr int H_   = 32;
constexpr int HD_  = 64;
constexpr int KV_  = 8;
constexpr int REP_ = 4;

// ---------------- bf16 helpers (bit-exact RNE, no backend cvt dependence) --

DEVINL __bf16 f2bf(float f) {
  union { float f; unsigned u; } v; v.f = f;
  unsigned r = v.u + 0x7FFFu + ((v.u >> 16) & 1u);
  unsigned short hs = (unsigned short)(r >> 16);
  __bf16 o; __builtin_memcpy(&o, &hs, 2); return o;
}
DEVINL float bf2f(__bf16 h) {
  unsigned short hs; __builtin_memcpy(&hs, &h, 2);
  unsigned u = ((unsigned)hs) << 16;
  float f; __builtin_memcpy(&f, &u, 4); return f;
}

DEVINL f32x8 wmma_bf16(bf16x16 a, bf16x16 b, f32x8 c) {
  return __builtin_amdgcn_wmma_f32_16x16x32_bf16(false, a, false, b,
                                                 (short)0, c, false, false);
}

// ---------------- CDNA5 async global->LDS (ASYNCcnt-tracked) ---------------
// Per-lane: LDS[vdst] = MEM[vaddr..vaddr+15].  LDS byte address = low 32 bits
// of the generic shared pointer (LDS aperture: addr[31:0] is the LDS offset).

DEVINL void async_ld_b128(void* lds, const void* g) {
  unsigned la = (unsigned)(unsigned long long)lds;
  asm volatile("global_load_async_to_lds_b128 %0, %1, off"
               :: "v"(la), "v"(g) : "memory");
}
DEVINL void wait_async0() { asm volatile("s_wait_asynccnt 0x0" ::: "memory"); }

// ---------------- WMMA fragment loaders (wave32 layouts per CDNA5 ISA) -----
// A 16x32 bf16: lane (l&15) = row M; elems 0..7 -> K = 8h+j ; 8..15 -> K = 16+8h+j
DEVINL bf16x16 a_frag_row(const __bf16* p, int lane) {
  int kh = ((lane >> 4) & 1) * 8;
  uint4 d0 = *(const uint4*)(p + kh);
  uint4 d1 = *(const uint4*)(p + 16 + kh);
  bf16x16 f;
  __builtin_memcpy(&f, &d0, 16);
  __builtin_memcpy(((char*)&f) + 16, &d1, 16);
  return f;
}
// B 32x16 bf16: lane (l&15) = col N; elems 0..7 -> K = 8h+j ; 8..15 -> K = 16+8h+j
DEVINL bf16x16 b_frag_row(const __bf16* p, int lane) {
  int kh = ((lane >> 4) & 1) * 8;
  uint4 d0 = *(const uint4*)(p + kh);
  uint4 d1 = *(const uint4*)(p + 16 + kh);
  bf16x16 f;
  __builtin_memcpy(&f, &d0, 16);
  __builtin_memcpy(((char*)&f) + 16, &d1, 16);
  return f;
}

// ---------------- kernel 0: f32 -> bf16 conversion -------------------------

__global__ __launch_bounds__(256) void cvt_kernel(const float* __restrict__ in,
                                                  __bf16* __restrict__ out,
                                                  long long n) {
  long long stride = (long long)gridDim.x * 256;
  for (long long i = (long long)blockIdx.x * 256 + threadIdx.x; i < n; i += stride)
    out[i] = f2bf(in[i]);
}

// ---------------- kernel 1: generic bf16 WMMA GEMM -------------------------
// C[M x N] = A[M x K] * W[K x N].  Block tile 128x128, 256 threads = 8 waves,
// wave tile 32x64 (2x4 accumulators).  Double-buffered LDS pipeline:
//   A staged row-major via global_load_async_to_lds_b128,
//   W staged TRANSPOSED ([n][k], pitch 40) so B fragments are 2x ds_load_b128.
// One barrier + one s_wait_asynccnt per K-step; next tile overlaps the WMMAs.

__global__ __launch_bounds__(256) void gemm_kernel(const __bf16* __restrict__ A,
                                                   const __bf16* __restrict__ W,
                                                   __bf16* __restrict__ Cb,
                                                   float*  __restrict__ Cf,
                                                   int M, int N, int K, int storef) {
  __shared__ __bf16 As[2 * 128 * 40];   // [buf][m][k]  (k pitch 40)
  __shared__ __bf16 Bst[2 * 128 * 40];  // [buf][n][k]  (k pitch 40, transposed)

  const int tid  = threadIdx.x;
  const int wave = tid >> 5, lane = tid & 31;
  const int hf   = lane >> 4, ln = lane & 15;
  const int bM   = blockIdx.y * 128, bN = blockIdx.x * 128;
  const int wm   = (wave >> 1) * 32, wn = (wave & 1) * 64;

  f32x8 acc[2][4] = {};

  const int ar = tid >> 1, ac = (tid & 1) * 16;  // A coop: 128 rows x 2 thr
  const int br = tid >> 3, bc = (tid & 7) * 16;  // W coop: 32 k-rows x 8 thr
  const __bf16* ag = A + (size_t)(bM + ar) * K + ac;
  const __bf16* bg = W + (size_t)br * N + bN + bc;

  auto stage = [&](int buf) {
    __bf16* as = As  + buf * (128 * 40);
    __bf16* bs = Bst + buf * (128 * 40);
    async_ld_b128(as + ar * 40 + ac,     ag);
    async_ld_b128(as + ar * 40 + ac + 8, ag + 8);
    uint4 b0 = *(const uint4*)(bg);
    uint4 b1 = *(const uint4*)(bg + 8);
    ag += 32;
    bg += (size_t)32 * N;
    __builtin_prefetch((const void*)bg, 0, 1);  // global_prefetch_b8, tile i+2
    __bf16 tb[16];
    __builtin_memcpy(tb,     &b0, 16);
    __builtin_memcpy(tb + 8, &b1, 16);
#pragma unroll
    for (int e = 0; e < 16; ++e) bs[(bc + e) * 40 + br] = tb[e];
  };

  stage(0);
  for (int kk = 0; kk < K; kk += 32) {
    const int cur = (kk >> 5) & 1;
    wait_async0();    // this wave's async fills for `cur` have landed
    __syncthreads();  // all waves staged `cur`; all reads of `cur^1` done
    if (kk + 32 < K) stage(cur ^ 1);  // overlap next fill with compute

    const __bf16* as = As  + cur * (128 * 40);
    const __bf16* bs = Bst + cur * (128 * 40);
    bf16x16 af0 = a_frag_row(as + (wm + ln) * 40, lane);
    bf16x16 af1 = a_frag_row(as + (wm + 16 + ln) * 40, lane);
#pragma unroll
    for (int tn = 0; tn < 4; ++tn) {
      bf16x16 wfrag = b_frag_row(bs + (wn + tn * 16 + ln) * 40, lane);
      acc[0][tn] = wmma_bf16(af0, wfrag, acc[0][tn]);
      acc[1][tn] = wmma_bf16(af1, wfrag, acc[1][tn]);
    }
  }

#pragma unroll
  for (int ti = 0; ti < 2; ++ti)
#pragma unroll
    for (int tn = 0; tn < 4; ++tn) {
      int rb = bM + wm + ti * 16, cb = bN + wn + tn * 16;
#pragma unroll
      for (int i = 0; i < 8; ++i) {
        size_t off = (size_t)(rb + i + 8 * hf) * N + cb + ln;
        if (storef) Cf[off] = acc[ti][tn][i];
        else        Cb[off] = f2bf(acc[ti][tn][i]);
      }
    }
}

// ---------------- kernel 2: RoPE on Q and K (bf16 in-place) ----------------

__global__ __launch_bounds__(256) void rope_kernel(__bf16* __restrict__ q,
                                                   __bf16* __restrict__ k,
                                                   const float* __restrict__ cosb,
                                                   const float* __restrict__ sinb) {
  const int PAIRS = HD_ / 2;  // 32
  const long long total = (long long)B_ * S_ * (H_ + KV_) * PAIRS;
  long long stride = (long long)gridDim.x * 256;
  for (long long idx = (long long)blockIdx.x * 256 + threadIdx.x; idx < total;
       idx += stride) {
    int p = (int)(idx % PAIRS);
    long long t = idx / PAIRS;
    int head = (int)(t % (H_ + KV_)); t /= (H_ + KV_);
    int s = (int)(t % S_);
    int b = (int)(t / S_);
    float c  = cosb[s * PAIRS + p];
    float sn = sinb[s * PAIRS + p];
    __bf16* base;
    if (head < H_) base = q + ((size_t)(b * S_ + s) * H_ + head) * HD_ + 2 * p;
    else           base = k + ((size_t)(b * S_ + s) * KV_ + (head - H_)) * HD_ + 2 * p;
    float t1 = bf2f(base[0]), t2 = bf2f(base[1]);
    base[0] = f2bf(t1 * c - t2 * sn);
    base[1] = f2bf(t1 * sn + t2 * c);
  }
}

// ---------------- kernel 3: flash attention (causal, GQA) ------------------
// grid = (S/64, B*H); block = 128 (4 waves).  Wave w owns 16 query rows.
// Double-buffered KV staging: K tile async-to-LDS row-major, V tile staged
// transposed [d][k].  One barrier + one s_wait_asynccnt per 32-wide KV step.

__global__ __launch_bounds__(128) void flash_kernel(const __bf16* __restrict__ q,
                                                    const __bf16* __restrict__ k,
                                                    const __bf16* __restrict__ v,
                                                    __bf16* __restrict__ o) {
  __shared__ __bf16 Ks[2 * 32 * 72];   // [buf][kpos][d]  (d pitch 72)
  __shared__ __bf16 Vst[2 * 64 * 40];  // [buf][d][kpos]  (k pitch 40)
  __shared__ __bf16 Ps[4 * 16 * 40];   // per-wave P transpose buffer

  const int tid  = threadIdx.x;
  const int wave = tid >> 5, lane = tid & 31;
  const int hf   = lane >> 4, ln = lane & 15;
  const int q0blk = blockIdx.x * 64;
  const int bh = blockIdx.y;
  const int h  = bh % H_;
  const int b  = bh / H_;
  const int g  = h / REP_;            // kv head
  const int qrow = q0blk + wave * 16; // wave's 16 rows

  const __bf16* qrowp = q + ((size_t)(b * S_ + qrow + ln) * H_ + h) * HD_;
  bf16x16 aq0 = a_frag_row(qrowp, lane);       // d = 0..31
  bf16x16 aq1 = a_frag_row(qrowp + 32, lane);  // d = 32..63

  f32x8 O[4] = {};
  float mrow[8], lrow[8];
#pragma unroll
  for (int i = 0; i < 8; ++i) { mrow[i] = -3.0e38f; lrow[i] = 0.f; }
  __bf16* Pw = Ps + wave * 16 * 40;

  auto stageF = [&](int buf, int kv) {
    __bf16* ks = Ks  + buf * (32 * 72);
    __bf16* vs = Vst + buf * (64 * 40);
    int r = tid >> 2, c0 = (tid & 3) * 16;
    size_t gidx = ((size_t)(b * S_ + kv + r) * KV_ + g) * HD_ + c0;
    async_ld_b128(ks + r * 72 + c0,     k + gidx);
    async_ld_b128(ks + r * 72 + c0 + 8, k + gidx + 8);
    uint4 dv0 = *(const uint4*)(v + gidx);
    uint4 dv1 = *(const uint4*)(v + gidx + 8);
    __bf16 tv[16];
    __builtin_memcpy(tv,     &dv0, 16);
    __builtin_memcpy(tv + 8, &dv1, 16);
#pragma unroll
    for (int e = 0; e < 16; ++e) vs[(c0 + e) * 40 + r] = tv[e];
  };

  const int kv_end = q0blk + 64;  // uniform trip count for whole block
  stageF(0, 0);
  for (int kv = 0; kv < kv_end; kv += 32) {
    const int cur = (kv >> 5) & 1;
    wait_async0();
    __syncthreads();
    if (kv + 32 < kv_end) stageF(cur ^ 1, kv + 32);

    const __bf16* ksb = Ks  + cur * (32 * 72);
    const __bf16* vsb = Vst + cur * (64 * 40);

    // scores: S = Q(16x64) . K^T(64x32)   -> two 16x16 f32 tiles
    f32x8 sc[2] = {};
#pragma unroll
    for (int t = 0; t < 2; ++t) {
      const __bf16* kp = ksb + (t * 16 + ln) * 72;
      bf16x16 bk0 = b_frag_row(kp, lane);       // d 0..31
      bf16x16 bk1 = b_frag_row(kp + 32, lane);  // d 32..63
      sc[t] = wmma_bf16(aq0, bk0, sc[t]);
      sc[t] = wmma_bf16(aq1, bk1, sc[t]);
    }

    // online softmax: scale, causal mask, row stats via shfl over 16-lane halves
#pragma unroll
    for (int i = 0; i < 8; ++i) {
      int qp = qrow + i + 8 * hf;  // this slot's query row
      float s0 = sc[0][i] * 0.125f + ((kv + ln      > qp) ? -1.0e9f : 0.f);
      float s1 = sc[1][i] * 0.125f + ((kv + 16 + ln > qp) ? -1.0e9f : 0.f);
      float mx = fmaxf(s0, s1);
#pragma unroll
      for (int mms = 1; mms < 16; mms <<= 1) mx = fmaxf(mx, __shfl_xor(mx, mms, 32));
      float mnew = fmaxf(mrow[i], mx);
      float fac  = __expf(mrow[i] - mnew);
      float p0 = __expf(s0 - mnew);
      float p1 = __expf(s1 - mnew);
      float ps = p0 + p1;
#pragma unroll
      for (int mms = 1; mms < 16; mms <<= 1) ps += __shfl_xor(ps, mms, 32);
      lrow[i] = lrow[i] * fac + ps;
      mrow[i] = mnew;
      O[0][i] *= fac; O[1][i] *= fac; O[2][i] *= fac; O[3][i] *= fac;
      // C-layout -> LDS (row-major 16x32 P tile, per-wave private region)
      Pw[(i + 8 * hf) * 40 + ln]      = f2bf(p0);
      Pw[(i + 8 * hf) * 40 + 16 + ln] = f2bf(p1);
    }
    // wave-local DS ordering before re-reading P in A-fragment layout
    asm volatile("s_wait_dscnt 0x0" ::: "memory");

    bf16x16 ap = a_frag_row(Pw + ln * 40, lane);  // P as 16x32 A fragment
#pragma unroll
    for (int tn = 0; tn < 4; ++tn) {
      bf16x16 bv = b_frag_row(vsb + (tn * 16 + ln) * 40, lane);
      O[tn] = wmma_bf16(ap, bv, O[tn]);
    }
  }

  // epilogue: normalize and store attention output (b, s, h, d) as bf16
#pragma unroll
  for (int tn = 0; tn < 4; ++tn)
#pragma unroll
    for (int i = 0; i < 8; ++i) {
      int qp = qrow + i + 8 * hf;
      float val = O[tn][i] / lrow[i];
      o[((size_t)(b * S_ + qp) * H_ + h) * HD_ + tn * 16 + ln] = f2bf(val);
    }
}

// ---------------- launch ----------------------------------------------------

extern "C" void kernel_launch(void* const* d_in, const int* in_sizes, int n_in,
                              void* d_out, int out_size, void* d_ws, size_t ws_size,
                              hipStream_t stream) {
  const float* x    = (const float*)d_in[0];
  const float* cosb = (const float*)d_in[1];
  const float* sinb = (const float*)d_in[2];
  // d_in[3] = mask (causal; recomputed inline), d_in[8] = start_pos (0)
  const float* wq = (const float*)d_in[4];
  const float* wk = (const float*)d_in[5];
  const float* wv = (const float*)d_in[6];
  const float* wo = (const float*)d_in[7];
  float* out = (float*)d_out;

  char* ws = (char*)d_ws;
  const size_t MB = 1ull << 20;
  __bf16* xb  = (__bf16*)(ws + 0);        // 16 MB  (B*S*E)
  __bf16* wqb = (__bf16*)(ws + 16 * MB);  //  8 MB
  __bf16* wkb = (__bf16*)(ws + 24 * MB);  //  2 MB
  __bf16* wvb = (__bf16*)(ws + 26 * MB);  //  2 MB
  __bf16* wob = (__bf16*)(ws + 28 * MB);  //  8 MB
  __bf16* qb  = (__bf16*)(ws + 36 * MB);  // 16 MB
  __bf16* kb  = (__bf16*)(ws + 52 * MB);  //  4 MB
  __bf16* vb  = (__bf16*)(ws + 56 * MB);  //  4 MB
  __bf16* ab  = (__bf16*)(ws + 60 * MB);  // 16 MB  -> 76 MB total

  cvt_kernel<<<2048, 256, 0, stream>>>(x,  xb,  (long long)B_ * S_ * E_);
  cvt_kernel<<<2048, 256, 0, stream>>>(wq, wqb, (long long)E_ * H_ * HD_);
  cvt_kernel<<<512,  256, 0, stream>>>(wk, wkb, (long long)E_ * KV_ * HD_);
  cvt_kernel<<<512,  256, 0, stream>>>(wv, wvb, (long long)E_ * KV_ * HD_);
  cvt_kernel<<<2048, 256, 0, stream>>>(wo, wob, (long long)H_ * HD_ * E_);

  const int M = B_ * S_;  // 4096
  dim3 blk(256);
  // Q = x @ wq ; K = x @ wk ; V = x @ wv   (bf16 out)
  gemm_kernel<<<dim3((H_ * HD_) / 128, M / 128), blk, 0, stream>>>(
      xb, wqb, qb, nullptr, M, H_ * HD_, E_, 0);
  gemm_kernel<<<dim3((KV_ * HD_) / 128, M / 128), blk, 0, stream>>>(
      xb, wkb, kb, nullptr, M, KV_ * HD_, E_, 0);
  gemm_kernel<<<dim3((KV_ * HD_) / 128, M / 128), blk, 0, stream>>>(
      xb, wvb, vb, nullptr, M, KV_ * HD_, E_, 0);

  rope_kernel<<<2048, 256, 0, stream>>>(qb, kb, cosb, sinb);

  flash_kernel<<<dim3(S_ / 64, B_ * H_), dim3(128), 0, stream>>>(qb, kb, vb, ab);

  // out = attn @ wo  (f32 out)
  gemm_kernel<<<dim3(E_ / 128, M / 128), blk, 0, stream>>>(
      ab, wob, nullptr, out, M, E_, H_ * HD_, 1);
  (void)in_sizes; (void)n_in; (void)out_size; (void)ws_size;
}
